// LinearAttention_1743756722387
// MI455X (gfx1250) — matrix operand order repaired and checked by reading, not compiled, not run on hardware
//
#include <hip/hip_runtime.h>

typedef __attribute__((ext_vector_type(16))) _Float16 v16h;
typedef __attribute__((ext_vector_type(8)))  _Float16 v8h;
typedef __attribute__((ext_vector_type(8)))  float    v8f;
typedef int v4i __attribute__((vector_size(16)));

#define DEVINL __device__ __forceinline__
#define AS1 __attribute__((address_space(1)))
#define AS3 __attribute__((address_space(3)))

constexpr int Bn  = 16;    // batch
constexpr int Cc  = 256;   // channels
constexpr int Nn  = 4096;  // H*W
constexpr int NHn = 4;     // heads
constexpr int HDn = 64;    // head dim
constexpr int Gg  = 32;    // groupnorm groups

// ---------------- workspace layout (bytes) ----------------
constexpr size_t OFF_H16    = 0;                                          // [B][C][N] f16
constexpr size_t OFF_QKVW16 = OFF_H16    + (size_t)Bn * Cc * Nn * 2;      // [3C][C] f16
constexpr size_t OFF_OUTW16 = OFF_QKVW16 + (size_t)3 * Cc * Cc * 2;       // [C][C] f16
constexpr size_t OFF_QKV32  = OFF_OUTW16 + (size_t)Cc * Cc * 2;           // [B][3C][N] f32
constexpr size_t OFF_Q16    = OFF_QKV32  + (size_t)Bn * 3 * Cc * Nn * 4;  // [B][C][N] f16
constexpr size_t OFF_K16    = OFF_Q16    + (size_t)Bn * Cc * Nn * 2;
constexpr size_t OFF_V16    = OFF_K16    + (size_t)Bn * Cc * Nn * 2;
constexpr size_t OFF_CTX16  = OFF_V16    + (size_t)Bn * Cc * Nn * 2;      // [B*NH][e=64][d=64] f16
constexpr size_t OFF_ATTN16 = OFF_CTX16  + (size_t)Bn * NHn * HDn * HDn * 2; // [B][C][N] f16

// ---------------- CDNA5 async-to-LDS staging (guarded; falls back to VGPR path) ----
#if defined(__AMDGCN__) && __has_builtin(__builtin_amdgcn_global_load_async_to_lds_b128)
#define USE_ASYNC_LDS 1
#else
#define USE_ASYNC_LDS 0
#endif

// 16-byte tile stage: global -> LDS (GLOBAL_LOAD_ASYNC_TO_LDS_B128, ASYNCcnt)
DEVINL void stage16B(const _Float16* g, _Float16* l) {
#if USE_ASYNC_LDS
  __builtin_amdgcn_global_load_async_to_lds_b128((AS1 v4i*)g, (AS3 v4i*)l, 0, 0);
#else
  *(v8h*)l = *(const v8h*)g;
#endif
}

DEVINL void async_stage_wait() {
#if USE_ASYNC_LDS
#if __has_builtin(__builtin_amdgcn_s_wait_asynccnt)
  __builtin_amdgcn_s_wait_asynccnt(0);
#else
  asm volatile("s_wait_asynccnt 0" ::: "memory");
#endif
#endif
}

// ---------------- WMMA fragment helpers ----------------
// A (16x32 f16, M x K): lane m = L%16; lanes<16 -> K 0..15,
// lanes>=16 -> K 16..31.  Tile stored row-major [m][k0..31].
DEVINL v16h load_a_frag(const _Float16* p, int ld, int lane) {
  int m  = lane & 15;
  int kb = (lane < 16) ? 0 : 16;
  union { v16h v; v8h h[2]; } u;
  u.h[0] = *(const v8h*)(p + m * ld + kb);
  u.h[1] = *(const v8h*)(p + m * ld + kb + 8);
  return u.v;
}
// B (32x16 f16, K x N): lane n = L%16; lanes<16 -> K 0..15, lanes>=16 -> K 16..31.
// Tile stored transposed [n][k0..31] so each lane reads 16 contiguous halves.
DEVINL v16h load_b_frag(const _Float16* p, int ld, int lane) {
  int n  = lane & 15;
  int kb = (lane < 16) ? 0 : 16;
  union { v16h v; v8h h[2]; } u;
  u.h[0] = *(const v8h*)(p + n * ld + kb);
  u.h[1] = *(const v8h*)(p + n * ld + kb + 8);
  return u.v;
}
DEVINL v8f wmma32(v16h a, v16h b, v8f c) {
  return __builtin_amdgcn_wmma_f32_16x16x32_f16(false, a, false, b, (short)0, c,
                                                false, false);
}

// ---------------- utility kernels ----------------
__global__ void f32_to_f16_kernel(const float* __restrict__ in,
                                  _Float16* __restrict__ out, int n) {
  int i = blockIdx.x * blockDim.x + threadIdx.x;
  if (i < n) out[i] = (_Float16)in[i];
}

// GroupNorm: one block per (batch, group); 8 channels * 4096 spatial each.
__global__ void groupnorm_kernel(const float* __restrict__ x,
                                 const float* __restrict__ w,
                                 const float* __restrict__ b,
                                 _Float16* __restrict__ h16) {
  const int CPG   = Cc / Gg;      // 8
  const int ELEMS = CPG * Nn;     // 32768
  int bg = blockIdx.x;
  int batch = bg / Gg, g = bg % Gg;
  const float* xp = x + ((size_t)batch * Cc + (size_t)g * CPG) * Nn;

  float s = 0.f, ss = 0.f;
  for (int i = threadIdx.x; i < ELEMS; i += blockDim.x) {
    float v = xp[i];
    s += v; ss += v * v;
  }
  for (int off = 16; off > 0; off >>= 1) {
    s  += __shfl_down(s, off);
    ss += __shfl_down(ss, off);
  }
  __shared__ float sred[16];
  int wid = threadIdx.x >> 5, lid = threadIdx.x & 31;
  if (lid == 0) { sred[wid] = s; sred[8 + wid] = ss; }
  __syncthreads();
  if (threadIdx.x == 0) {
    float a = 0.f, c = 0.f;
    for (int i = 0; i < 8; ++i) { a += sred[i]; c += sred[8 + i]; }
    sred[0] = a; sred[8] = c;
  }
  __syncthreads();
  float mean = sred[0] * (1.f / ELEMS);
  float var  = sred[8] * (1.f / ELEMS) - mean * mean;
  float rstd = rsqrtf(var + 1e-5f);

  _Float16* hp = h16 + ((size_t)batch * Cc + (size_t)g * CPG) * Nn;
  for (int i = threadIdx.x; i < ELEMS; i += blockDim.x) {
    int c = g * CPG + i / Nn;
    hp[i] = (_Float16)((xp[i] - mean) * rstd * w[c] + b[c]);
  }
}

// ---------------- generic batched GEMM: C[b] = A x B[b] (+bias)(+resid) ----------------
// A: [M][K] f16 (shared). B: [b][K][Nn] f16. Out: [b][M][Nn] f32.
// MODE 0: out = acc + bias[m]            (qkv projection)
// MODE 1: out = acc + bias[m] + resid    (final projection + residual)
template <int MODE>
__global__ void gemm_f16_kernel(const _Float16* __restrict__ A,
                                const _Float16* __restrict__ Bm,
                                const float* __restrict__ bias,
                                const float* __restrict__ resid,
                                float* __restrict__ Out, int M, int K) {
  constexpr int BM = 128, BN = 128, BK = 32, LD = BK + 8;
  int batch = blockIdx.z;
  int m0 = blockIdx.y * BM;
  int n0 = blockIdx.x * BN;
  const _Float16* Bp = Bm + (size_t)batch * K * Nn;
  float* Op = Out + (size_t)batch * M * Nn;
  const float* Rp = (MODE == 1) ? (resid + (size_t)batch * M * Nn) : nullptr;

  __shared__ _Float16 sA[BM][LD];
  __shared__ _Float16 sB[BN][LD];

  int t = threadIdx.x, w = t >> 5, lane = t & 31;
  int wm = (w >> 1) * 32, wn = (w & 1) * 64;   // wave tile 32x64

  v8f acc[2][4] = {};

  for (int k0 = 0; k0 < K; k0 += BK) {
    __syncthreads();
#pragma unroll
    for (int it = 0; it < 2; ++it) {           // A tile: 128x32 halves, async to LDS
      int cid = t + it * 256;
      int row = cid >> 2, c8 = (cid & 3) * 8;
      stage16B(A + (size_t)(m0 + row) * K + k0 + c8, &sA[row][c8]);
    }
#pragma unroll
    for (int it = 0; it < 2; ++it) {           // B tile: 32x128, transposed into LDS
      int cid = t + it * 256;
      int kk = cid >> 4, n8 = (cid & 15) * 8;
      v8h tmp = *(const v8h*)(Bp + (size_t)(k0 + kk) * Nn + n0 + n8);
#pragma unroll
      for (int j = 0; j < 8; ++j) sB[n8 + j][kk] = tmp[j];
    }
    if (k0 + BK < K) {                          // prefetch next K tile
      __builtin_prefetch(A + (size_t)(m0 + (t & 127)) * K + k0 + BK, 0, 1);
      __builtin_prefetch(Bp + (size_t)(k0 + BK + (t & 31)) * Nn + n0, 0, 1);
    }
    async_stage_wait();
    __syncthreads();

    v16h af[2], bf[4];
#pragma unroll
    for (int i = 0; i < 2; ++i) af[i] = load_a_frag(&sA[wm + i * 16][0], LD, lane);
#pragma unroll
    for (int j = 0; j < 4; ++j) bf[j] = load_b_frag(&sB[wn + j * 16][0], LD, lane);
#pragma unroll
    for (int i = 0; i < 2; ++i)
#pragma unroll
      for (int j = 0; j < 4; ++j) acc[i][j] = wmma32(af[i], bf[j], acc[i][j]);
  }

#pragma unroll
  for (int i = 0; i < 2; ++i)
#pragma unroll
    for (int j = 0; j < 4; ++j)
#pragma unroll
      for (int r = 0; r < 8; ++r) {
        int mi = m0 + wm + i * 16 + r + ((lane < 16) ? 0 : 8);
        int ni = n0 + wn + j * 16 + (lane & 15);
        float val = acc[i][j][r] + bias[mi];
        if (MODE == 1) val += Rp[(size_t)mi * Nn + ni];
        Op[(size_t)mi * Nn + ni] = val;
      }
}

// ---------------- softmax over spatial N (q) ----------------
__global__ void softmax_rows_q_kernel(const float* __restrict__ qkv32,
                                      _Float16* __restrict__ q16) {
  int row = blockIdx.x;                // b*C + c
  int b = row / Cc, c = row % Cc;
  const float* p = qkv32 + ((size_t)b * 3 * Cc + c) * Nn;
  _Float16* o = q16 + (size_t)row * Nn;

  __shared__ float sm[8], ssum[8];
  int wid = threadIdx.x >> 5, lid = threadIdx.x & 31;

  float m = -3.4e38f;
  for (int i = threadIdx.x; i < Nn; i += blockDim.x) m = fmaxf(m, p[i]);
  for (int off = 16; off > 0; off >>= 1) m = fmaxf(m, __shfl_down(m, off));
  if (lid == 0) sm[wid] = m;
  __syncthreads();
  if (threadIdx.x == 0) {
    float a = sm[0];
    for (int i = 1; i < 8; ++i) a = fmaxf(a, sm[i]);
    sm[0] = a;
  }
  __syncthreads();
  m = sm[0];

  float s = 0.f;
  for (int i = threadIdx.x; i < Nn; i += blockDim.x) s += __expf(p[i] - m);
  for (int off = 16; off > 0; off >>= 1) s += __shfl_down(s, off);
  if (lid == 0) ssum[wid] = s;
  __syncthreads();
  if (threadIdx.x == 0) {
    float a = 0.f;
    for (int i = 0; i < 8; ++i) a += ssum[i];
    ssum[0] = a;
  }
  __syncthreads();
  float inv = 1.f / ssum[0];
  for (int i = threadIdx.x; i < Nn; i += blockDim.x)
    o[i] = (_Float16)(__expf(p[i] - m) * inv);
}

// ---------------- softmax over head-dim d (k), one thread per (b,h,n) ----------------
__global__ void softmax_cols_k_kernel(const float* __restrict__ qkv32,
                                      _Float16* __restrict__ k16) {
  int gid = blockIdx.x * blockDim.x + threadIdx.x;   // over B*NH*N
  if (gid >= Bn * NHn * Nn) return;
  int n = gid % Nn, bh = gid / Nn;
  int b = bh / NHn, h = bh % NHn;
  const float* p = qkv32 + ((size_t)b * 3 * Cc + Cc + (size_t)h * HDn) * Nn + n;
  _Float16* o = k16 + ((size_t)b * Cc + (size_t)h * HDn) * Nn + n;

  float m = -3.4e38f;
  for (int d = 0; d < HDn; ++d) m = fmaxf(m, p[(size_t)d * Nn]);
  float s = 0.f;
  for (int d = 0; d < HDn; ++d) s += __expf(p[(size_t)d * Nn] - m);
  float inv = 1.f / s;
  for (int d = 0; d < HDn; ++d)
    o[(size_t)d * Nn] = (_Float16)(__expf(p[(size_t)d * Nn] - m) * inv);
}

// ---------------- v cast to f16 ----------------
__global__ void cast_v_kernel(const float* __restrict__ qkv32,
                              _Float16* __restrict__ v16) {
  size_t i = (size_t)blockIdx.x * blockDim.x + threadIdx.x;
  size_t per = (size_t)Cc * Nn;
  if (i >= (size_t)Bn * per) return;
  size_t b = i / per, r = i % per;
  v16[i] = (_Float16)qkv32[b * 3 * per + 2 * per + r];
}

// ---------------- context = K (64xN) x V^T (Nx64) per (b,h); stores ctx^T f16 --------
// BK=64: both tiles contiguous rows -> pure async staging, 4 WMMA per barrier round.
__global__ void ctx_gemm_kernel(const _Float16* __restrict__ k16,
                                const _Float16* __restrict__ v16,
                                _Float16* __restrict__ ctx16) {
  constexpr int BK = 64, LD = BK + 8;    // 72 halves = 144B rows (16B aligned)
  int bh = blockIdx.x;
  int b = bh / NHn, h = bh % NHn;
  const _Float16* Ap = k16 + ((size_t)b * Cc + (size_t)h * HDn) * Nn;  // rows d
  const _Float16* Vp = v16 + ((size_t)b * Cc + (size_t)h * HDn) * Nn;  // rows e

  __shared__ _Float16 sA[64][LD];
  __shared__ _Float16 sB[64][LD];   // [e][kk] -> contiguous rows of V

  int t = threadIdx.x, w = t >> 5, lane = t & 31;
  int wm = (w >> 1) * 16, wn = (w & 1) * 32;   // wave tile 16x32

  v8f acc[2] = {};

  for (int k0 = 0; k0 < Nn; k0 += BK) {
    __syncthreads();
#pragma unroll
    for (int it = 0; it < 2; ++it) {     // 64 rows x 64 halves each tile
      int cid = t + it * 256;
      int row = cid >> 3, c8 = (cid & 7) * 8;
      stage16B(Ap + (size_t)row * Nn + k0 + c8, &sA[row][c8]);
      stage16B(Vp + (size_t)row * Nn + k0 + c8, &sB[row][c8]);
    }
    async_stage_wait();
    __syncthreads();

#pragma unroll
    for (int ks = 0; ks < 2; ++ks) {     // two 16x16x32 sub-steps
      v16h af  = load_a_frag(&sA[wm][ks * 32], LD, lane);
      v16h bf0 = load_b_frag(&sB[wn][ks * 32], LD, lane);
      v16h bf1 = load_b_frag(&sB[wn + 16][ks * 32], LD, lane);
      acc[0] = wmma32(af, bf0, acc[0]);
      acc[1] = wmma32(af, bf1, acc[1]);
    }
  }

#pragma unroll
  for (int j = 0; j < 2; ++j)
#pragma unroll
    for (int r = 0; r < 8; ++r) {
      int d = wm + r + ((lane < 16) ? 0 : 8);
      int e = wn + j * 16 + (lane & 15);
      // store transposed: ctx16[bh][e][d]
      ctx16[((size_t)bh * HDn + e) * HDn + d] = (_Float16)acc[j][r];
    }
}

// ---------------- out = ctx^T (64x64) x Q (64xN) per (b,h), f16 result ----------------
__global__ void attn_out_gemm_kernel(const _Float16* __restrict__ ctx16,
                                     const _Float16* __restrict__ q16,
                                     _Float16* __restrict__ attn16) {
  constexpr int BN = 128, BK = 32, LD = BK + 8;
  int bh = blockIdx.y;
  int n0 = blockIdx.x * BN;
  int b = bh / NHn, h = bh % NHn;
  const _Float16* Ap = ctx16 + (size_t)bh * HDn * HDn;                  // [e][d]
  const _Float16* Bq = q16 + ((size_t)b * Cc + (size_t)h * HDn) * Nn;   // rows d

  __shared__ _Float16 sA[64][LD];
  __shared__ _Float16 sB[BN][LD];

  int t = threadIdx.x, w = t >> 5, lane = t & 31;
  int wm = (w >> 1) * 16, wn = (w & 1) * 64;   // wave tile 16x64

  v8f acc[4] = {};

  for (int k0 = 0; k0 < HDn; k0 += BK) {   // 2 iterations
    __syncthreads();
    {
      int row = t >> 2, c8 = (t & 3) * 8;
      stage16B(Ap + (size_t)row * HDn + k0 + c8, &sA[row][c8]);
    }
#pragma unroll
    for (int it = 0; it < 2; ++it) {       // B tile transposed into LDS
      int cid = t + it * 256;
      int kk = cid >> 4, n8 = (cid & 15) * 8;
      v8h tmp = *(const v8h*)(Bq + (size_t)(k0 + kk) * Nn + n0 + n8);
#pragma unroll
      for (int j = 0; j < 8; ++j) sB[n8 + j][kk] = tmp[j];
    }
    async_stage_wait();
    __syncthreads();

    v16h af = load_a_frag(&sA[wm][0], LD, lane);
#pragma unroll
    for (int j = 0; j < 4; ++j) {
      v16h bf = load_b_frag(&sB[wn + j * 16][0], LD, lane);
      acc[j] = wmma32(af, bf, acc[j]);
    }
  }

#pragma unroll
  for (int j = 0; j < 4; ++j)
#pragma unroll
    for (int r = 0; r < 8; ++r) {
      int e = wm + r + ((lane < 16) ? 0 : 8);
      int n = n0 + wn + j * 16 + (lane & 15);
      attn16[((size_t)b * Cc + (size_t)h * HDn + e) * Nn + n] = (_Float16)acc[j][r];
    }
}

// ---------------- launcher ----------------
extern "C" void kernel_launch(void* const* d_in, const int* in_sizes, int n_in,
                              void* d_out, int out_size, void* d_ws, size_t ws_size,
                              hipStream_t stream) {
  const float* x     = (const float*)d_in[0];
  const float* gn_w  = (const float*)d_in[1];
  const float* gn_b  = (const float*)d_in[2];
  const float* qkv_w = (const float*)d_in[3];
  const float* qkv_b = (const float*)d_in[4];
  const float* out_w = (const float*)d_in[5];
  const float* out_b = (const float*)d_in[6];

  char* ws = (char*)d_ws;
  _Float16* h16    = (_Float16*)(ws + OFF_H16);
  _Float16* qkvw16 = (_Float16*)(ws + OFF_QKVW16);
  _Float16* outw16 = (_Float16*)(ws + OFF_OUTW16);
  float*    qkv32  = (float*)   (ws + OFF_QKV32);
  _Float16* q16    = (_Float16*)(ws + OFF_Q16);
  _Float16* k16    = (_Float16*)(ws + OFF_K16);
  _Float16* v16    = (_Float16*)(ws + OFF_V16);
  _Float16* ctx16  = (_Float16*)(ws + OFF_CTX16);
  _Float16* attn16 = (_Float16*)(ws + OFF_ATTN16);

  // weights to f16
  f32_to_f16_kernel<<<(3 * Cc * Cc + 255) / 256, 256, 0, stream>>>(qkv_w, qkvw16, 3 * Cc * Cc);
  f32_to_f16_kernel<<<(Cc * Cc + 255) / 256, 256, 0, stream>>>(out_w, outw16, Cc * Cc);

  // group norm -> h16
  groupnorm_kernel<<<Bn * Gg, 256, 0, stream>>>(x, gn_w, gn_b, h16);

  // qkv = qkv_w @ h + qkv_b  (M=768, K=256, N=4096, 16 batches)
  gemm_f16_kernel<0><<<dim3(Nn / 128, (3 * Cc) / 128, Bn), 256, 0, stream>>>(
      qkvw16, h16, qkv_b, nullptr, qkv32, 3 * Cc, Cc);

  // softmaxes + v cast
  softmax_rows_q_kernel<<<Bn * Cc, 256, 0, stream>>>(qkv32, q16);
  softmax_cols_k_kernel<<<(Bn * NHn * Nn) / 256, 256, 0, stream>>>(qkv32, k16);
  cast_v_kernel<<<(Bn * Cc * Nn) / 256, 256, 0, stream>>>(qkv32, v16);

  // context (per head) and attention output
  ctx_gemm_kernel<<<Bn * NHn, 256, 0, stream>>>(k16, v16, ctx16);
  attn_out_gemm_kernel<<<dim3(Nn / 128, Bn * NHn), 256, 0, stream>>>(ctx16, q16, attn16);

  // final projection + bias + residual (M=256, K=256, N=4096, 16 batches)
  gemm_f16_kernel<1><<<dim3(Nn / 128, Cc / 128, Bn), 256, 0, stream>>>(
      outw16, attn16, out_b, x, (float*)d_out, Cc, Cc);
}
